// Net_69715909149412
// MI455X (gfx1250) — matrix operand order repaired and checked by reading, not compiled, not run on hardware
//
#include <hip/hip_runtime.h>
#include <hip/hip_bf16.h>

typedef __attribute__((ext_vector_type(16))) _Float16 v16h;
typedef __attribute__((ext_vector_type(4)))  _Float16 v4h;
typedef __attribute__((ext_vector_type(8)))  float    v8f;

#define B_SZ 2048
#define T_SZ 90
#define A_SZ 4
#define D_SZ 256
#define H_SZ 10
#define O_SZ 2
#define HP   16                 // H padded to WMMA N=16
#define ROWS (B_SZ * T_SZ)      // 184320 flattened (b,t) rows
#define WPB  4                  // waves (tiles) per block in kernel 1

// ---------------------------------------------------------------------------
// Kernel 1: antenna fuse + hidden linear via WMMA.
//   sn[(b*T+t)*16 + h] = b_hid[h] + sum_d ( (b_ant + sum_a w_ant[a]*x[b,t,a,d]) * w_hid[h,d] )
// One wave owns a 16-row tile of the [ROWS, D] fused matrix.
// ---------------------------------------------------------------------------
__global__ __launch_bounds__(128) void snn_fuse_gemm_kernel(
    const float* __restrict__ x,      // [B,T,A,D]
    const float* __restrict__ w_ant,  // [A]
    const float* __restrict__ b_ant,  // [1]
    const float* __restrict__ w_hid,  // [H,D]
    const float* __restrict__ b_hid,  // [H]
    float* __restrict__ sn)           // [ROWS, HP]
{
    __shared__ __align__(16) _Float16 sWhid[HP * D_SZ];          // 8 KB, block-shared
    __shared__ __align__(16) _Float16 sFused[WPB][16 * D_SZ];    // 4 x 8 KB, wave-private

    const int tid  = threadIdx.x;
    const int wv   = tid >> 5;
    const int lane = tid & 31;

    // Stage w_hid as f16 (zero-pad rows h >= 10).
    for (int i = tid; i < HP * D_SZ; i += 128) {
        const int h = i >> 8;
        const int d = i & (D_SZ - 1);
        const float v = (h < H_SZ) ? w_hid[h * D_SZ + d] : 0.0f;
        sWhid[i] = (_Float16)v;
    }
    __syncthreads();

    const float wa0 = w_ant[0], wa1 = w_ant[1], wa2 = w_ant[2], wa3 = w_ant[3];
    const float ba  = b_ant[0];

    const long tile = (long)blockIdx.x * WPB + wv;
    const long r0   = tile * 16;                   // first flattened row of this tile

    // ---- Phase 1: stream x, antenna-combine, deposit f16 tile in LDS ----
    for (int m = 0; m < 16; ++m) {
        const float* rowp = x + (r0 + m) * (A_SZ * D_SZ);
        if (m + 1 < 16)   // keep the HBM stream ahead of us (global_prefetch_b8)
            __builtin_prefetch(x + (r0 + m + 1) * (A_SZ * D_SZ) + lane * 32, 0, 0);
#pragma unroll
        for (int i = 0; i < 2; ++i) {
            const int d = (i * 32 + lane) * 4;     // coalesced 512B per wave
            const float4 x0 = *(const float4*)(rowp + 0 * D_SZ + d);
            const float4 x1 = *(const float4*)(rowp + 1 * D_SZ + d);
            const float4 x2 = *(const float4*)(rowp + 2 * D_SZ + d);
            const float4 x3 = *(const float4*)(rowp + 3 * D_SZ + d);
            v4h f;
            f.x = (_Float16)(ba + wa0 * x0.x + wa1 * x1.x + wa2 * x2.x + wa3 * x3.x);
            f.y = (_Float16)(ba + wa0 * x0.y + wa1 * x1.y + wa2 * x2.y + wa3 * x3.y);
            f.z = (_Float16)(ba + wa0 * x0.z + wa1 * x1.z + wa2 * x2.z + wa3 * x3.z);
            f.w = (_Float16)(ba + wa0 * x0.w + wa1 * x1.w + wa2 * x2.w + wa3 * x3.w);
            *(v4h*)&sFused[wv][m * D_SZ + d] = f;
        }
    }
    // Tile is wave-private: lanes of this wave issue LDS ops in lockstep, so a
    // DS-counter drain (no workgroup barrier) makes the stores visible.
    asm volatile("s_wait_dscnt 0" ::: "memory");

    // ---- Phase 2: 8x v_wmma_f32_16x16x32_f16 over K = 256 ----
    // A fragment (16x32 f16): lane l -> row M = l%16; K base 0 (lanes 0-15) or 8
    // (lanes 16-31); VGPR0-3 = K base..base+7, VGPR4-7 = base+16..base+23.
    // B fragment (32x16 f16): lane l -> col N = l%16; same K layout as A (base 0
    // or 8; VGPR0-3 = base..base+7, VGPR4-7 = base+16..base+23).
    const int mA    = lane & 15;
    const int koffA = (lane < 16) ? 0 : 8;
    const int nB    = lane & 15;
    const int koffB = (lane < 16) ? 0 : 8;

    const float hb = (nB < H_SZ) ? b_hid[nB] : 0.0f;
    v8f c;
#pragma unroll
    for (int r = 0; r < 8; ++r) c[r] = hb;

    union Frag { uint4 u4[2]; v16h h; };

#pragma unroll
    for (int ck = 0; ck < 8; ++ck) {
        const int kc = ck * 32;
        Frag fa, fb;
        fa.u4[0] = *(const uint4*)&sFused[wv][mA * D_SZ + kc + koffA];
        fa.u4[1] = *(const uint4*)&sFused[wv][mA * D_SZ + kc + koffA + 16];
        fb.u4[0] = *(const uint4*)&sWhid[nB * D_SZ + kc + koffB];
        fb.u4[1] = *(const uint4*)&sWhid[nB * D_SZ + kc + koffB + 16];
        c = __builtin_amdgcn_wmma_f32_16x16x32_f16(
                /*neg_a=*/false, fa.h, /*neg_b=*/false, fb.h,
                /*c_mod=*/(short)0, c, /*reuse_a=*/false, /*reuse_b=*/false);
    }

    // ---- Phase 3: scatter C (16x16 f32) to sn[row, h] ----
    // VGPR r: lanes 0-15 -> M=r, lanes 16-31 -> M=r+8; col N = lane%16.
    const int mrow = (lane < 16) ? 0 : 8;
#pragma unroll
    for (int r = 0; r < 8; ++r)
        sn[(r0 + mrow + r) * HP + nB] = c[r];
}

// ---------------------------------------------------------------------------
// Kernel 2: LIF scan over T (zero-reset from previous mem), time fuse,
// output linear, softmax. 16 threads (one per padded h) per batch.
// ---------------------------------------------------------------------------
__global__ __launch_bounds__(256) void snn_scan_out_kernel(
    const float* __restrict__ sn,      // [ROWS, HP]
    const float* __restrict__ w_time,  // [T]
    const float* __restrict__ b_time,  // [1]
    const float* __restrict__ w_out,   // [O,H]
    const float* __restrict__ b_out,   // [O]
    float* __restrict__ out)           // [B,O]
{
    __shared__ float sft[16][HP];

    const int tid = threadIdx.x;
    const int bl  = tid >> 4;          // batch within block (0..15)
    const int h   = tid & 15;          // padded hidden index
    const long b  = (long)blockIdx.x * 16 + bl;

    const float* rowp = sn + (b * T_SZ) * HP + h;

    float mem = 0.0f, acc = 0.0f;
    for (int t = 0; t < T_SZ; ++t) {
        const float inp  = rowp[t * HP];
        const float keep = (mem > 1.0f) ? 0.0f : 1.0f;   // reset from PREVIOUS mem
        mem = fmaf(0.95f, mem, inp) * keep;              // BETA*mem + inp, zero-reset
        if (mem > 1.0f) acc += w_time[t];                // spk * w_time[t]
    }
    sft[bl][h] = acc + b_time[0];
    __syncthreads();

    if (tid < 16) {
        const int bb = tid;
        float s0 = b_out[0], s1 = b_out[1];
#pragma unroll
        for (int hh = 0; hh < H_SZ; ++hh) {
            const float f = sft[bb][hh];
            s0 = fmaf(f, w_out[0 * H_SZ + hh], s0);
            s1 = fmaf(f, w_out[1 * H_SZ + hh], s1);
        }
        const float mx  = fmaxf(s0, s1);
        const float e0  = __expf(s0 - mx);
        const float e1  = __expf(s1 - mx);
        const float inv = 1.0f / (e0 + e1);
        const long bg   = (long)blockIdx.x * 16 + bb;
        out[bg * 2 + 0] = e0 * inv;
        out[bg * 2 + 1] = e1 * inv;
    }
}

// ---------------------------------------------------------------------------
extern "C" void kernel_launch(void* const* d_in, const int* in_sizes, int n_in,
                              void* d_out, int out_size, void* d_ws, size_t ws_size,
                              hipStream_t stream) {
    const float* x      = (const float*)d_in[0];
    const float* w_ant  = (const float*)d_in[1];
    const float* b_ant  = (const float*)d_in[2];
    const float* w_hid  = (const float*)d_in[3];
    const float* b_hid  = (const float*)d_in[4];
    const float* w_time = (const float*)d_in[5];
    const float* b_time = (const float*)d_in[6];
    const float* w_out  = (const float*)d_in[7];
    const float* b_out  = (const float*)d_in[8];

    float* sn = (float*)d_ws;   // ROWS * HP * 4 = 11.8 MB scratch (fits in L2)

    // 184320 rows / (16 rows * 4 waves) = 2880 blocks of 128 threads.
    snn_fuse_gemm_kernel<<<ROWS / (16 * WPB), 128, 0, stream>>>(
        x, w_ant, b_ant, w_hid, b_hid, sn);

    // 2048 batches / 16 per block = 128 blocks of 256 threads.
    snn_scan_out_kernel<<<B_SZ / 16, 256, 0, stream>>>(
        sn, w_time, b_time, w_out, b_out, (float*)d_out);
}
